// ConfidenceBasedCE_12524124636028
// MI455X (gfx1250) — matrix operand order repaired and checked by reading, not compiled, not run on hardware
//
#include <hip/hip_runtime.h>
#include <hip/hip_bf16.h>

typedef __attribute__((ext_vector_type(16))) _Float16 v16h;
typedef __attribute__((ext_vector_type(8)))  float    v8f;
typedef __attribute__((ext_vector_type(4)))  unsigned u32x4;
typedef __attribute__((ext_vector_type(8)))  unsigned u32x8;

#if defined(__has_builtin)
#if __has_builtin(__builtin_amdgcn_tensor_load_to_lds)
#define HAVE_TDM_BUILTIN 1
#endif
#endif

// Inline-asm TDM path (used when the clang builtin is absent).
#ifndef USE_TDM_ASM
#define USE_TDM_ASM 1
#endif

static constexpr int C_ = 10;   // classes
static constexpr int K_ = 50;   // neighbors
static constexpr float CT1 = 0.15f;
static constexpr float CT2 = 0.1f;

// ---------------------------------------------------------------- init / zero
__global__ __launch_bounds__(64)
void k_init(int* cnt0, int* cnt1, int* counts, float* lossacc) {
  int t = threadIdx.x;
  if (t == 0) { *cnt0 = 0; *cnt1 = 0; *lossacc = 0.0f; }
  if (t < 16) counts[t] = 0;
}

__global__ __launch_bounds__(256)
void k_zero_out(float* out, int n) {
  int i = blockIdx.x * 256 + threadIdx.x;
  if (i < n) out[i] = 0.0f;
}

// -------------------------------------------------- stage 1: confidence mask
__global__ __launch_bounds__(256)
void k_stage1(const float* __restrict__ aweak, int* __restrict__ idx0,
              int* __restrict__ cnt0, int B) {
  int b = blockIdx.x * 256 + threadIdx.x;
  if (b >= B) return;
  const float* row = aweak + (size_t)b * C_;
  float x[C_], mx = -1e30f;
  #pragma unroll
  for (int c = 0; c < C_; ++c) { x[c] = row[c]; mx = fmaxf(mx, x[c]); }
  float sm = 0.0f;
  #pragma unroll
  for (int c = 0; c < C_; ++c) sm += __expf(x[c] - mx);
  float maxprob = 1.0f / sm;                 // exp(mx-mx)/sum
  if (maxprob > CT1) {
    int p = atomicAdd(cnt0, 1);
    idx0[p] = b;
  }
}

// -------------------------------------- stage 2: WMMA attention, q_hat, beta
// One wave per selected row. The contiguous 500-float neighbor block is DMA'd
// into LDS via the Tensor Data Mover (builtin if present, else inline asm),
// then the WMMA A-fill and the softmax pass read LDS with wide vector loads.
// Scores via v_wmma_f32_16x16x32_f16 with B = per-row vector broadcast over
// all 16 columns, so every accumulator column carries the tile's 16 scores.
__global__ __launch_bounds__(32)
void k_stage2(const float* __restrict__ anchors,
              const float* __restrict__ neighbors,
              const float* __restrict__ att,
              const float* __restrict__ proj,
              const int* __restrict__ idx0,
              const int* __restrict__ cnt0,
              float* __restrict__ beta_out) {
  __shared__ __attribute__((aligned(16))) float nbs[512];  // K_*C_ = 500 floats
  __shared__ float vv[16];
  __shared__ float scores[64];
  __shared__ float qh[16];
  __shared__ float smax_s, ssum_s;

  int i = blockIdx.x;
  int n0 = *cnt0;
  if (i >= n0) return;                        // wave-uniform; EXEC stays full
  int lane = threadIdx.x;
  int b = idx0[i];
  const float* nb = neighbors + (size_t)b * K_ * C_;

#if defined(HAVE_TDM_BUILTIN) || USE_TDM_ASM
  {
    // 1-D D# descriptor: 500 x 4B contiguous elements  ->  LDS
    unsigned lds_off = (unsigned)(uintptr_t)&nbs[0];   // low 32 = LDS offset
    unsigned long long ga = (unsigned long long)(uintptr_t)nb;
    const unsigned NE = (unsigned)(K_ * C_);           // 500
    u32x4 g0;
    g0[0] = 1u;                                        // count=1, user mode
    g0[1] = lds_off;                                   // lds_addr
    g0[2] = (unsigned)(ga & 0xFFFFFFFFull);            // global_addr[31:0]
    g0[3] = (unsigned)((ga >> 32) & 0x1FFFFFFull)      // global_addr[56:32]
          | (2u << 30);                                // type = 2 ("image")
    u32x8 g1;
    g1[0] = 2u << 16;                                  // data_size = 4 bytes
    g1[1] = (NE & 0xFFFFu) << 16;                      // tensor_dim0 lo16
    g1[2] = (NE >> 16) | (1u << 16);                   // tensor_dim0 hi|dim1=1
    g1[3] = NE << 16;                                  // tile_dim0 = 500
    g1[4] = 0u;                                        // tile_dim1/2 unused
    g1[5] = NE;                                        // tensor_dim0_stride
    g1[6] = 0u;
    g1[7] = 0u;
#if defined(HAVE_TDM_BUILTIN)
    typedef __attribute__((ext_vector_type(4))) int i32x4;
    typedef __attribute__((ext_vector_type(8))) int i32x8;
    i32x4 z4 = {};
#if __clang_major__ >= 23
    i32x8 z8 = {};
    __builtin_amdgcn_tensor_load_to_lds((u32x4)g0, (i32x8)(g1), z4, z4, z8, 0);
#else
    __builtin_amdgcn_tensor_load_to_lds((u32x4)g0, (i32x8)(g1), z4, z4, 0);
#endif
#else
    // 2-group form: VADDR2/VADDR3 = NULL (tensor up to 2D)
    asm volatile("tensor_load_to_lds %0, %1"
                 :: "s"(g0), "s"(g1)
                 : "memory");
#endif
#if __has_builtin(__builtin_amdgcn_s_wait_tensorcnt)
    __builtin_amdgcn_s_wait_tensorcnt(0);
#else
    asm volatile("s_wait_tensorcnt 0x0" ::: "memory");
#endif
  }
#else
  // fallback: cooperative vectorized stage (2000B block is 16B-aligned)
  for (int t = lane; t < (K_ * C_ / 4); t += 32)
    ((float4*)nbs)[t] = ((const float4*)nb)[t];
#endif

  // v[k] = att[k] * diag(proj)[k]^2 * anchors[b,k]
  if (lane < C_) {
    float d = proj[lane * C_ + lane];
    vv[lane] = att[lane] * d * d * anchors[(size_t)b * C_ + lane];
  }
  if (lane < 16) qh[lane] = 0.0f;
  __syncthreads();

  int half = lane >> 4;
  int m    = lane & 15;

  // B operand (32x16 f16): lanes 0-15 slots 0..7 = K0..7 (v[0..7]);
  // lanes 16-31 slots 0..1 = K8..9 (v[8..9]); all other slots zero.
  v16h bmat = {};
  if (half == 0) {
    #pragma unroll
    for (int s = 0; s < 8; ++s) bmat[s] = (_Float16)vv[s];
  } else {
    bmat[0] = (_Float16)vv[8]; bmat[1] = (_Float16)vv[9];
  }

  for (int jt = 0; jt < 4; ++jt) {
    int j0 = jt * 16;
    int j  = j0 + m;
    // A operand (16x32 f16): lanes 0-15 slots 0..7 = K0..7, lanes 16-31
    // slots 0..1 = K8..9; all other slots are zero padding (K >= 10).
    v16h amat = {};
    if (j < K_) {
      const float* rp = &nbs[j * C_];          // 8B-aligned (40*j bytes)
      if (half == 0) {
        float2 p0 = ((const float2*)rp)[0];
        float2 p1 = ((const float2*)rp)[1];
        float2 p2 = ((const float2*)rp)[2];
        float2 p3 = ((const float2*)rp)[3];
        amat[0] = (_Float16)p0.x; amat[1] = (_Float16)p0.y;
        amat[2] = (_Float16)p1.x; amat[3] = (_Float16)p1.y;
        amat[4] = (_Float16)p2.x; amat[5] = (_Float16)p2.y;
        amat[6] = (_Float16)p3.x; amat[7] = (_Float16)p3.y;
      } else {
        float2 p4 = ((const float2*)rp)[4];
        amat[0] = (_Float16)p4.x; amat[1] = (_Float16)p4.y;
      }
    }
    v8f acc = {};
    acc = __builtin_amdgcn_wmma_f32_16x16x32_f16(
        /*neg_a=*/false, amat, /*neg_b=*/false, bmat,
        /*c_mod=*/(short)0, acc, /*reuse_a=*/false, /*reuse_b=*/false);
    // D layout: VGPR r, lanes 0-15 -> M=r; lanes 16-31 -> M=8+r (cols equal)
    if (lane == 0) {
      #pragma unroll
      for (int r = 0; r < 8; ++r) {
        float sc = acc[r];
        scores[j0 + r] = (sc > 0.0f) ? sc : 0.01f * sc;   // leaky_relu
      }
    }
    if (lane == 16) {
      #pragma unroll
      for (int r = 0; r < 8; ++r) {
        float sc = acc[r];
        scores[j0 + 8 + r] = (sc > 0.0f) ? sc : 0.01f * sc;
      }
    }
  }
  __syncthreads();

  if (lane == 0) {
    float mx = -1e30f;
    for (int j = 0; j < K_; ++j) mx = fmaxf(mx, scores[j]);
    float sm = 0.0f;
    for (int j = 0; j < K_; ++j) sm += __expf(scores[j] - mx);
    smax_s = mx; ssum_s = sm;
  }
  __syncthreads();
  float smax = smax_s, inv_ssum = 1.0f / ssum_s;

  // q_hat[c] = sum_j softmax(scores)[j] * softmax(neighbors[b,j,:])[c]
  for (int j = lane; j < K_; j += 32) {
    float x[C_], mx = -1e30f;
    #pragma unroll
    for (int c = 0; c < C_; ++c) { x[c] = nbs[j * C_ + c]; mx = fmaxf(mx, x[c]); }
    float sm = 0.0f;
    #pragma unroll
    for (int c = 0; c < C_; ++c) { x[c] = __expf(x[c] - mx); sm += x[c]; }
    float w = __expf(scores[j] - smax) * inv_ssum / sm;
    #pragma unroll
    for (int c = 0; c < C_; ++c) atomicAdd(&qh[c], w * x[c]);  // ds_add_f32
  }
  __syncthreads();

  if (lane == 0) {
    const float* arow = anchors + (size_t)b * C_;
    float x[C_], mx = -1e30f;
    #pragma unroll
    for (int c = 0; c < C_; ++c) { x[c] = arow[c]; mx = fmaxf(mx, x[c]); }
    float sm = 0.0f;
    #pragma unroll
    for (int c = 0; c < C_; ++c) { x[c] = __expf(x[c] - mx); sm += x[c]; }
    float bsum = 0.0f;
    #pragma unroll
    for (int c = 0; c < C_; ++c) {
      float d = x[c] / sm - qh[c];
      bsum += d * d;
    }
    beta_out[i] = bsum;
  }
}

// -------------------------- stage 3: exact k-th smallest beta (radix select)
__global__ __launch_bounds__(1024)
void k_stage3(const float* __restrict__ beta, const int* __restrict__ cnt0,
              float* __restrict__ tau_out) {
  __shared__ int hist[2048];
  __shared__ unsigned sh_prefix, sh_mask;
  __shared__ int sh_rank;
  int t = threadIdx.x;
  int n0 = *cnt0;
  if (n0 == 0) { if (t == 0) *tau_out = 1.0f; return; }
  int k = n0 / 2; if (k < 1) k = 1;

  if (t == 0) { sh_prefix = 0u; sh_mask = 0u; sh_rank = k; }
  __syncthreads();

  const int shifts[3] = {21, 10, 0};
  const int nbits[3]  = {11, 11, 10};
  for (int lvl = 0; lvl < 3; ++lvl) {
    int nb = 1 << nbits[lvl];
    for (int h = t; h < nb; h += 1024) hist[h] = 0;
    __syncthreads();
    unsigned prefix = sh_prefix, mask = sh_mask;
    for (int i = t; i < n0; i += 1024) {
      unsigned u = __float_as_uint(beta[i]);      // beta >= 0: bit order = order
      if ((u & mask) == prefix) {
        unsigned bin = (u >> shifts[lvl]) & (unsigned)(nb - 1);
        atomicAdd(&hist[bin], 1);
      }
    }
    __syncthreads();
    if (t == 0) {
      int rank = sh_rank, cum = 0, sb = nb - 1;
      for (int bn = 0; bn < nb; ++bn) {
        if (cum + hist[bn] >= rank) { sb = bn; break; }
        cum += hist[bn];
      }
      sh_rank = sh_rank - cum;
      sh_prefix = sh_prefix | ((unsigned)sb << shifts[lvl]);
      sh_mask   = sh_mask   | ((unsigned)(nb - 1) << shifts[lvl]);
    }
    __syncthreads();
  }
  if (t == 0) {
    float kth = __uint_as_float(sh_prefix);
    *tau_out = kth * 2.718281828459045f;           // / exp(-1)
  }
}

// ------------------- stage 4: alpha, sharpened q, second mask + class counts
__global__ __launch_bounds__(256)
void k_stage4(const float* __restrict__ aweak, const int* __restrict__ idx0,
              const int* __restrict__ cnt0, const float* __restrict__ beta,
              const float* __restrict__ tau_p, const int* __restrict__ labels,
              float* __restrict__ qrows, int* __restrict__ sel,
              int* __restrict__ tgt, int* __restrict__ labs,
              int* __restrict__ cnt1, int* __restrict__ counts) {
  int i = blockIdx.x * 256 + threadIdx.x;
  int n0 = *cnt0;
  if (i >= n0) return;
  int b = idx0[i];
  float tau = *tau_p;
  float alpha = -logf(beta[i] / tau);

  const float* row = aweak + (size_t)b * C_;
  float x[C_], mx = -1e30f;
  #pragma unroll
  for (int c = 0; c < C_; ++c) { x[c] = row[c]; mx = fmaxf(mx, x[c]); }
  float sm = 0.0f;
  #pragma unroll
  for (int c = 0; c < C_; ++c) { x[c] = __expf(x[c] - mx); sm += x[c]; }
  float q[C_];
  if (alpha > 1.0f) {
    float qs = 0.0f;
    #pragma unroll
    for (int c = 0; c < C_; ++c) { q[c] = powf(x[c] / sm, alpha); qs += q[c]; }
    float inv = 1.0f / qs;
    #pragma unroll
    for (int c = 0; c < C_; ++c) q[c] *= inv;
  } else {
    #pragma unroll
    for (int c = 0; c < C_; ++c) q[c] = x[c] / sm;
  }
  int amax = 0; float mp = q[0];
  #pragma unroll
  for (int c = 1; c < C_; ++c) if (q[c] > mp) { mp = q[c]; amax = c; }
  #pragma unroll
  for (int c = 0; c < C_; ++c) qrows[(size_t)i * C_ + c] = q[c];

  if (mp > CT2) {
    int p = atomicAdd(cnt1, 1);
    sel[p]  = i;
    tgt[p]  = amax;
    labs[p] = labels[b];
    atomicAdd(&counts[amax], 1);
  }
}

// ----------------------------------------- stage 5a: class-balancing weights
__global__ __launch_bounds__(32)
void k_stage5a(const int* __restrict__ counts, const int* __restrict__ cnt1,
               float* __restrict__ wavg) {
  if (threadIdx.x != 0) return;
  int n = *cnt1; if (n < 1) n = 1;
  float w[C_]; float sum = 0.0f;
  #pragma unroll
  for (int c = 0; c < C_; ++c) {
    float freq = (float)counts[c] / (float)n;
    w[c] = (counts[c] > 0) ? (1.0f / logf(1.02f + freq)) : 1.0f;
    sum += w[c];
  }
  float mean = sum / (float)C_;
  #pragma unroll
  for (int c = 0; c < C_; ++c) wavg[c] = w[c] / sum * mean;
}

// ----------------------------- stage 5b: weighted CE partials + int outputs
__global__ __launch_bounds__(256)
void k_stage5b(const float* __restrict__ astrong, const int* __restrict__ idx0,
               const int* __restrict__ sel, const int* __restrict__ tgt,
               const int* __restrict__ labs, const int* __restrict__ cnt1,
               const float* __restrict__ qrows, const float* __restrict__ wavg,
               float* __restrict__ lossacc, float* __restrict__ out,
               int n_ref, int out_size) {
  int pos = blockIdx.x * 256 + threadIdx.x;
  int n = *cnt1;
  if (pos >= n) return;
  int i = sel[pos];
  int b = idx0[i];
  const float* row = astrong + (size_t)b * C_;
  float x[C_], mx = -1e30f;
  #pragma unroll
  for (int c = 0; c < C_; ++c) { x[c] = row[c]; mx = fmaxf(mx, x[c]); }
  float sm = 0.0f;
  #pragma unroll
  for (int c = 0; c < C_; ++c) sm += __expf(x[c] - mx);
  float lse = logf(sm) + mx;
  float s = 0.0f;
  #pragma unroll
  for (int c = 0; c < C_; ++c)
    s += wavg[c] * qrows[(size_t)i * C_ + c] * (x[c] - lse);
  atomicAdd(lossacc, s);                       // global_atomic_add_f32
  if (pos < n_ref) {
    if (1 + pos < out_size)         out[1 + pos]         = (float)tgt[pos];
    if (1 + n_ref + pos < out_size) out[1 + n_ref + pos] = (float)labs[pos];
  }
}

__global__ __launch_bounds__(32)
void k_stage5c(const float* __restrict__ lossacc, const int* __restrict__ cnt1,
               float* __restrict__ out, int n_ref, int out_size) {
  if (threadIdx.x != 0) return;
  int n = *cnt1;
  out[0] = (n > 0) ? (-(*lossacc) / (float)n) : 0.0f;
  int npos = 1 + 2 * n_ref;
  if (npos < out_size) out[npos] = (float)n;
}

// --------------------------------------------------------------------- host
extern "C" void kernel_launch(void* const* d_in, const int* in_sizes, int n_in,
                              void* d_out, int out_size, void* d_ws, size_t ws_size,
                              hipStream_t stream) {
  const float* aweak   = (const float*)d_in[0];
  const float* astrong = (const float*)d_in[1];
  const float* anchors = (const float*)d_in[2];
  const float* neigh   = (const float*)d_in[3];
  const int*   labels  = (const int*)d_in[4];
  const float* att     = (const float*)d_in[6];
  const float* proj    = (const float*)d_in[7];

  const int B = in_sizes[0] / C_;

  // workspace carve-up (floats/ints, 4B units)
  int*   cnt0    = (int*)d_ws;
  int*   cnt1    = cnt0 + 1;
  int*   counts  = cnt0 + 2;           // 16 ints
  float* tau     = (float*)(cnt0 + 18);
  float* lossacc = tau + 1;
  float* wavg    = lossacc + 1;        // 16 floats
  float* base    = (float*)d_ws + 64;
  int*   idx0  = (int*)base;                           // B
  float* beta  = base + B;                             // B
  float* qrows = beta + B;                             // B*C
  int*   sel   = (int*)(qrows + (size_t)B * C_);       // B
  int*   tgt   = sel + B;                              // B
  int*   labs  = tgt + B;                              // B

  int n_ref = (out_size - 2) / 2; if (n_ref < 0) n_ref = 0;
  float* out = (float*)d_out;

  k_init<<<1, 64, 0, stream>>>(cnt0, cnt1, counts, lossacc);
  k_zero_out<<<(out_size + 255) / 256, 256, 0, stream>>>(out, out_size);
  k_stage1<<<(B + 255) / 256, 256, 0, stream>>>(aweak, idx0, cnt0, B);
  k_stage2<<<B, 32, 0, stream>>>(anchors, neigh, att, proj, idx0, cnt0, beta);
  k_stage3<<<1, 1024, 0, stream>>>(beta, cnt0, tau);
  k_stage4<<<(B + 255) / 256, 256, 0, stream>>>(aweak, idx0, cnt0, beta, tau,
                                                labels, qrows, sel, tgt, labs,
                                                cnt1, counts);
  k_stage5a<<<1, 32, 0, stream>>>(counts, cnt1, wavg);
  k_stage5b<<<(B + 255) / 256, 256, 0, stream>>>(astrong, idx0, sel, tgt, labs,
                                                 cnt1, qrows, wavg, lossacc,
                                                 out, n_ref, out_size);
  k_stage5c<<<1, 32, 0, stream>>>(lossacc, cnt1, out, n_ref, out_size);
}
